// W8A8Linear_7524782703138
// MI455X (gfx1250) — 8-bit matrix operand order repaired and checked by reading, compiled, not run on hardware
//
#include <hip/hip_runtime.h>
#include <stdint.h>

#define M_TOT 8192
#define N_TOT 4096
#define K_TOT 4096

#define BM 128
#define BN 128
#define BK 64
#define LSTRIDE 80
#define PLANE  (BM * LSTRIDE)
#define BUFSZ  (3 * PLANE)

typedef __attribute__((ext_vector_type(8))) int v8i;
typedef __attribute__((ext_vector_type(4))) int v4i;

#if defined(__AMDGCN__) && \
    __has_builtin(__builtin_amdgcn_global_load_async_to_lds_b128) && \
    __has_builtin(__builtin_amdgcn_s_wait_asynccnt)
#define USE_ASYNC 1
#else
#define USE_ASYNC 0
#endif

__global__ __launch_bounds__(256) void quantize_x_kernel(
    const float* __restrict__ x, const float* __restrict__ is_p,
    unsigned char* __restrict__ lo, char* __restrict__ hi)
{
    const float is = *is_p;
    size_t i = (size_t)blockIdx.x * 256 + threadIdx.x;
    float4 v = reinterpret_cast<const float4*>(x)[i];
    int q0 = __float2int_rn(v.x / is);
    int q1 = __float2int_rn(v.y / is);
    int q2 = __float2int_rn(v.z / is);
    int q3 = __float2int_rn(v.w / is);
    uchar4 l;
    l.x = (unsigned char)(q0 & 0xFF);
    l.y = (unsigned char)(q1 & 0xFF);
    l.z = (unsigned char)(q2 & 0xFF);
    l.w = (unsigned char)(q3 & 0xFF);
    char4 h;
    h.x = (char)(q0 >> 8);
    h.y = (char)(q1 >> 8);
    h.z = (char)(q2 >> 8);
    h.w = (char)(q3 >> 8);
    reinterpret_cast<uchar4*>(lo)[i] = l;
    reinterpret_cast<char4*>(hi)[i]  = h;
}

__global__ __launch_bounds__(256) void pack_w_kernel(
    const int* __restrict__ w, char* __restrict__ w8)
{
    size_t i = (size_t)blockIdx.x * 256 + threadIdx.x;
    int4 v = reinterpret_cast<const int4*>(w)[i];
    char4 c;
    c.x = (char)v.x; c.y = (char)v.y; c.z = (char)v.z; c.w = (char)v.w;
    reinterpret_cast<char4*>(w8)[i] = c;
}

#if USE_ASYNC
typedef __attribute__((address_space(1))) v4i* gas_v4i_ptr;
typedef __attribute__((address_space(3))) v4i* las_v4i_ptr;

__device__ __forceinline__ void async_cp16(const void* g, void* l)
{
    __builtin_amdgcn_global_load_async_to_lds_b128(
        (gas_v4i_ptr)g, (las_v4i_ptr)l, 0, 0);
}

__device__ __forceinline__ void issue_tile(
    unsigned char* buf,
    const unsigned char* __restrict__ Alo,
    const char*          __restrict__ Ahi,
    const char*          __restrict__ Bw,
    int m0, int n0, int kk, int lr, int lc)
{
    const size_t ga = (size_t)(m0 + lr) * K_TOT + (size_t)kk + lc;
    const size_t gb = (size_t)(n0 + lr) * K_TOT + (size_t)kk + lc;
    unsigned char* pa = buf + lr * LSTRIDE + lc;
    unsigned char* ph = pa + PLANE;
    unsigned char* pb = pa + 2 * PLANE;
    async_cp16(Alo + ga,      pa);
    async_cp16(Alo + ga + 16, pa + 16);
    async_cp16(Ahi + ga,      ph);
    async_cp16(Ahi + ga + 16, ph + 16);
    async_cp16(Bw  + gb,      pb);
    async_cp16(Bw  + gb + 16, pb + 16);
}
#endif

__device__ __forceinline__ void compute_step(
    const unsigned char* buf, int wm, int wn, int lh, int ll,
    v8i acc_lo[4][2], v8i acc_hi[4][2])
{
    const unsigned char* tAlo = buf;
    const unsigned char* tAhi = buf + PLANE;
    const unsigned char* tB   = buf + 2 * PLANE;

    v8i bfrag[2];
#pragma unroll
    for (int nt = 0; nt < 2; ++nt) {
        const unsigned char* cp =
            &tB[(wn * 32 + nt * 16 + ll) * LSTRIDE + lh * 16];
        uint4 t0 = *reinterpret_cast<const uint4*>(cp);
        uint4 t1 = *reinterpret_cast<const uint4*>(cp + 32);
        v8i b;
        b[0] = (int)t0.x; b[1] = (int)t0.y; b[2] = (int)t0.z; b[3] = (int)t0.w;
        b[4] = (int)t1.x; b[5] = (int)t1.y; b[6] = (int)t1.z; b[7] = (int)t1.w;
        bfrag[nt] = b;
    }

#pragma unroll
    for (int mt = 0; mt < 4; ++mt) {
        const int row = wm * 64 + mt * 16 + ll;
        const unsigned char* rpl = &tAlo[row * LSTRIDE + lh * 16];
        const unsigned char* rph = &tAhi[row * LSTRIDE + lh * 16];
        v8i alo, ahi;
#pragma unroll
        for (int j = 0; j < 2; ++j) {
            uint4 tl = *reinterpret_cast<const uint4*>(rpl + j * 32);
            uint4 th = *reinterpret_cast<const uint4*>(rph + j * 32);
            alo[4 * j]     = (int)tl.x;
            alo[4 * j + 1] = (int)tl.y;
            alo[4 * j + 2] = (int)tl.z;
            alo[4 * j + 3] = (int)tl.w;
            ahi[4 * j]     = (int)th.x;
            ahi[4 * j + 1] = (int)th.y;
            ahi[4 * j + 2] = (int)th.z;
            ahi[4 * j + 3] = (int)th.w;
        }
#pragma unroll
        for (int nt = 0; nt < 2; ++nt) {
            acc_lo[mt][nt] = __builtin_amdgcn_wmma_i32_16x16x64_iu8(
                false, alo, true, bfrag[nt], acc_lo[mt][nt], false, false);
            acc_hi[mt][nt] = __builtin_amdgcn_wmma_i32_16x16x64_iu8(
                true, ahi, true, bfrag[nt], acc_hi[mt][nt], false, false);
        }
    }
}

__global__ __launch_bounds__(256) void w8a8_gemm_kernel(
    const unsigned char* __restrict__ Alo,
    const char*          __restrict__ Ahi,
    const char*          __restrict__ Bw,
    const int*           __restrict__ bias,
    const float*         __restrict__ wscale,
    const float*         __restrict__ is_p,
    const float*         __restrict__ os_p,
    int*                 __restrict__ out)
{
    const int tid  = threadIdx.x;
    const int lane = tid & 31;
    const int wid  = tid >> 5;
    const int wm   = wid >> 2;
    const int wn   = wid & 3;
    const int lh   = lane >> 4;
    const int ll   = lane & 15;

    const int m0 = blockIdx.y * BM;
    const int n0 = blockIdx.x * BN;

    const int lr = tid >> 1;
    const int lc = (tid & 1) * 32;

    v8i acc_lo[4][2];
    v8i acc_hi[4][2];
#pragma unroll
    for (int mt = 0; mt < 4; ++mt)
#pragma unroll
        for (int nt = 0; nt < 2; ++nt) {
            v8i z = {};
            acc_lo[mt][nt] = z;
            acc_hi[mt][nt] = z;
        }

#if USE_ASYNC
    __shared__ __align__(16) unsigned char smem[2 * BUFSZ];

    issue_tile(smem, Alo, Ahi, Bw, m0, n0, 0, lr, lc);

    const int nk = K_TOT / BK;
    for (int i = 0; i < nk; ++i) {
        __builtin_amdgcn_s_wait_asynccnt(0);
        asm volatile("s_wait_dscnt 0x0\n\t"
                     "s_barrier_signal -1\n\t"
                     "s_barrier_wait -1" ::: "memory");
        if (i + 1 < nk)
            issue_tile(smem + ((i + 1) & 1) * BUFSZ,
                       Alo, Ahi, Bw, m0, n0, (i + 1) * BK, lr, lc);
        compute_step(smem + (i & 1) * BUFSZ, wm, wn, lh, ll, acc_lo, acc_hi);
    }
#else
    __shared__ __align__(16) unsigned char smem[BUFSZ];
    for (int kk = 0; kk < K_TOT; kk += BK) {
        __syncthreads();
        const size_t ga = (size_t)(m0 + lr) * K_TOT + (size_t)kk + lc;
        const size_t gb = (size_t)(n0 + lr) * K_TOT + (size_t)kk + lc;
        uint4 a0 = *reinterpret_cast<const uint4*>(Alo + ga);
        uint4 a1 = *reinterpret_cast<const uint4*>(Alo + ga + 16);
        uint4 h0 = *reinterpret_cast<const uint4*>(Ahi + ga);
        uint4 h1 = *reinterpret_cast<const uint4*>(Ahi + ga + 16);
        uint4 b0 = *reinterpret_cast<const uint4*>(Bw  + gb);
        uint4 b1 = *reinterpret_cast<const uint4*>(Bw  + gb + 16);
        unsigned char* pa = &smem[lr * LSTRIDE + lc];
        reinterpret_cast<uint4*>(pa)[0]             = a0;
        reinterpret_cast<uint4*>(pa)[1]             = a1;
        reinterpret_cast<uint4*>(pa + PLANE)[0]     = h0;
        reinterpret_cast<uint4*>(pa + PLANE)[1]     = h1;
        reinterpret_cast<uint4*>(pa + 2 * PLANE)[0] = b0;
        reinterpret_cast<uint4*>(pa + 2 * PLANE)[1] = b1;
        if (kk + BK < K_TOT) {
            __builtin_prefetch(Alo + ga + BK, 0, 0);
            __builtin_prefetch(Ahi + ga + BK, 0, 0);
            __builtin_prefetch(Bw  + gb + BK, 0, 0);
        }
        __syncthreads();
        compute_step(smem, wm, wn, lh, ll, acc_lo, acc_hi);
    }
#endif

    const float is = *is_p;
    const float os = *os_p;
#pragma unroll
    for (int mt = 0; mt < 4; ++mt) {
#pragma unroll
        for (int nt = 0; nt < 2; ++nt) {
            const int gn = n0 + wn * 32 + nt * 16 + ll;
            const float sc = is * wscale[gn];
            const float bz = (float)bias[gn];
            const int mbase = m0 + wm * 64 + mt * 16 + lh * 8;
#pragma unroll
            for (int e = 0; e < 8; ++e) {
                const int a = acc_lo[mt][nt][e] + (acc_hi[mt][nt][e] << 8);
                const float r = sc * (float)a + bz;
                out[(size_t)(mbase + e) * N_TOT + gn] = __float2int_rn(r / os);
            }
        }
    }
}

extern "C" void kernel_launch(void* const* d_in, const int* in_sizes, int n_in,
                              void* d_out, int out_size, void* d_ws, size_t ws_size,
                              hipStream_t stream)
{
    const float* x      = (const float*)d_in[0];
    const int*   weight = (const int*)d_in[1];
    const int*   bias   = (const int*)d_in[2];
    const float* wscale = (const float*)d_in[3];
    const float* is_p   = (const float*)d_in[4];
    const float* os_p   = (const float*)d_in[5];
    int*         out    = (int*)d_out;

    unsigned char* Alo = (unsigned char*)d_ws;
    char*          Ahi = (char*)d_ws + (size_t)M_TOT * K_TOT;
    char*          Bw  = (char*)d_ws + 2 * (size_t)M_TOT * K_TOT;

    quantize_x_kernel<<<(M_TOT * (size_t)K_TOT) / 4 / 256, 256, 0, stream>>>(
        x, is_p, Alo, Ahi);
    pack_w_kernel<<<(N_TOT * (size_t)K_TOT) / 4 / 256, 256, 0, stream>>>(
        weight, Bw);

    dim3 grid(N_TOT / BN, M_TOT / BM);
    w8a8_gemm_kernel<<<grid, 256, 0, stream>>>(
        Alo, Ahi, Bw, bias, wscale, is_p, os_p, out);
}
